// ExpertMLPsCapacityFactor_44805098832176
// MI455X (gfx1250) — matrix operand order repaired and checked by reading, not compiled, not run on hardware
//
#include <hip/hip_runtime.h>

// ---------------- problem constants ----------------
#define T_TOK 8192
#define NE    8
#define TOPK  2
#define HID   1024
#define INTER_ 4096
#define CAP   2560              // ceil(T*K*1.25/E)
#define ECAP  (NE*CAP)          // 20480

// ---------------- types ----------------
typedef __attribute__((ext_vector_type(16))) __bf16 v16bf;
typedef __attribute__((ext_vector_type(8)))  float  v8f;
typedef __attribute__((ext_vector_type(4)))  unsigned u32x4;
typedef __attribute__((ext_vector_type(2)))  unsigned u32x2;
typedef __attribute__((ext_vector_type(4)))  float    f32x4;

union FragBF { u32x4 u[2]; v16bf v; };   // 32B: one WMMA bf16 A/B operand per lane

#if __has_builtin(__builtin_amdgcn_global_load_async_to_lds_b128)
#define HAVE_ASYNC_LDS 1
#else
#define HAVE_ASYNC_LDS 0
#endif

static __device__ __forceinline__ unsigned short f2bf(float f) {
  unsigned u = __builtin_bit_cast(unsigned, f);
  u += 0x7FFFu + ((u >> 16) & 1u);      // round-to-nearest-even
  return (unsigned short)(u >> 16);
}
static __device__ __forceinline__ unsigned pack2bf(float a, float b) {
  return (unsigned)f2bf(a) | ((unsigned)f2bf(b) << 16);
}

#if HAVE_ASYNC_LDS
// Async Global->LDS 16B copy (GLOBAL_LOAD_ASYNC_TO_LDS_B128, ASYNCcnt).
// Builtin signature (from toolchain diagnostic): typed v4i pointers,
// param 1 = v4i addrspace(1)* (global src), param 2 = LDS dst, then imm, imm.
typedef int v4i_gcc __attribute__((vector_size(16)));
typedef v4i_gcc __attribute__((address_space(1))) as1_v4i;
typedef v4i_gcc __attribute__((address_space(3))) as3_v4i;
static __device__ __forceinline__ void async_copy16(const void* g, void* l) {
  __builtin_amdgcn_global_load_async_to_lds_b128(
      (as1_v4i*)g, (as3_v4i*)l, 0, 0);
}
#endif
static __device__ __forceinline__ void wait_async0() {
#if __has_builtin(__builtin_amdgcn_s_wait_asynccnt)
  __builtin_amdgcn_s_wait_asynccnt(0);
#else
  asm volatile("s_wait_asynccnt 0x0" ::: "memory");
#endif
}

// ---------------------------------------------------------------------------
// Kernel 1: fp32 weights -> bf16, re-blocked into WMMA B-fragment order.
// Block = 32(K) x 16(N). Within a block (1024B):
//   lane L = (n&15) + 16*((k>>4)&1) ; dword j = (k&15)>>1 ; lo/hi = k&1
// Block order: nb-major, kb inner  (blk = nb*(Kdim/32) + kb), expert-major.
// => a K-slice for one nb is contiguous; lane fragment read = 32B contiguous.
// ---------------------------------------------------------------------------
__global__ __launch_bounds__(256) void convert_weights(
    const float* __restrict__ src, unsigned* __restrict__ dst,
    int Kdim, int N, long totalDwords) {
  const int P  = (Kdim >> 1) * N;   // dwords per expert
  const int KB = Kdim >> 5;
  for (long d = blockIdx.x * (long)blockDim.x + threadIdx.x; d < totalDwords;
       d += (long)gridDim.x * blockDim.x) {
    int e = (int)(d / P);
    int r = (int)(d - (long)e * P);
    int blk = r >> 8, within = r & 255;
    int L = within >> 3, j = within & 7;
    int kb = blk % KB, nb = blk / KB;
    int n = nb * 16 + (L & 15);
    int k = kb * 32 + ((L >> 4) << 4) + (j << 1);
    const float* s = src + (long)e * Kdim * N;
    dst[d] = pack2bf(s[(long)k * N + n], s[(long)(k + 1) * N + n]);
  }
}

// ---------------------------------------------------------------------------
// Kernel 2: routing. Single block, 1024 threads, 8 tokens/thread.
// Packed 8x16-bit per-expert counts -> Hillis-Steele block scan (fields can't
// overflow: per-expert totals <= T = 8192 < 65536). Deterministic.
// ---------------------------------------------------------------------------
__global__ __launch_bounds__(1024) void routing_kernel(
    const int* __restrict__ eidx, const float* __restrict__ aff,
    int* __restrict__ slot2tok, int* __restrict__ tokslot,
    float* __restrict__ tokw) {
  const int t = threadIdx.x;
  for (int i = t; i < ECAP; i += 1024) slot2tok[i] = -1;

  __shared__ u32x4 scanBuf[1024];
  unsigned cnt[NE];
#pragma unroll
  for (int e = 0; e < NE; e++) cnt[e] = 0;
  const int tok0 = t * 8;
#pragma unroll
  for (int i = 0; i < 8; i++) {
    cnt[eidx[(tok0 + i) * 2 + 0] & 7]++;
    cnt[eidx[(tok0 + i) * 2 + 1] & 7]++;
  }
  u32x4 v;
  v[0] = cnt[0] | (cnt[1] << 16); v[1] = cnt[2] | (cnt[3] << 16);
  v[2] = cnt[4] | (cnt[5] << 16); v[3] = cnt[6] | (cnt[7] << 16);
  u32x4 incl = v;
  scanBuf[t] = incl;
  __syncthreads();
  for (int off = 1; off < 1024; off <<= 1) {
    u32x4 add = {0u, 0u, 0u, 0u};
    if (t >= off) add = scanBuf[t - off];
    __syncthreads();
    incl += add;
    scanBuf[t] = incl;
    __syncthreads();
  }
  u32x4 ex = incl - v;                   // exclusive; no inter-field borrow
  unsigned run[NE];
#pragma unroll
  for (int e = 0; e < 4; e++) { run[2 * e] = ex[e] & 0xFFFFu; run[2 * e + 1] = ex[e] >> 16; }

  for (int i = 0; i < 8; i++) {
    int tok = tok0 + i;
    int i0 = eidx[tok * 2 + 0] & 7, i1 = eidx[tok * 2 + 1] & 7;
    float a0 = aff[tok * NE + i0], a1 = aff[tok * NE + i1];
    float inv = 1.0f / (a0 + a1);
    int p0 = (int)run[i0]++;
    int p1 = (int)run[i1]++;
    bool k0 = p0 < CAP, k1 = p1 < CAP;
    tokslot[tok * 2 + 0] = k0 ? i0 * CAP + p0 : -1;
    tokslot[tok * 2 + 1] = k1 ? i1 * CAP + p1 : -1;
    tokw[tok * 2 + 0] = k0 ? a0 * inv : 0.0f;
    tokw[tok * 2 + 1] = k1 ? a1 * inv : 0.0f;
    if (k0) slot2tok[i0 * CAP + p0] = tok;
    if (k1) slot2tok[i1 * CAP + p1] = tok;
  }
}

// ---------------------------------------------------------------------------
// Kernel 3: dispatch tokens into per-expert bf16 buffers (row-major).
// ---------------------------------------------------------------------------
__global__ __launch_bounds__(256) void dispatch_kernel(
    const float* __restrict__ x, const int* __restrict__ slot2tok,
    u32x2* __restrict__ ebuf) {
  int row = blockIdx.x;
  int tok = slot2tok[row];
  int c = threadIdx.x;                  // 4 elements per thread
  u32x2 o = {0u, 0u};
  if (tok >= 0) {
    f32x4 vv = ((const f32x4*)(x + (long)tok * HID))[c];
    o[0] = pack2bf(vv[0], vv[1]);
    o[1] = pack2bf(vv[2], vv[3]);
  }
  ebuf[(long)row * (HID / 4) + c] = o;
}

// ---------------------------------------------------------------------------
// Kernel 4: fused gate_up GEMM + SiLU*up -> bf16 intermediate.
// WG = 256 thr (8 waves, 2x4). Wave tile: M=64, N=32 for BOTH gate and up.
// WG tile: 128 x 128 (gate cols nBase.., up cols nBase+INTER..).
// K-step = 32 (one v_wmma_f32_16x16x32_bf16 per tile pair). Double-buffered
// LDS; tiles streamed with GLOBAL_LOAD_ASYNC_TO_LDS_B128 (ASYNCcnt) so the
// next tile's DMA overlaps the WMMAs; one barrier per K iteration.
// A fragment (contiguous k = 16h + s, as B): lane(m=L&15, h=L>>4) reads two
// contiguous 16B chunks at k = 16h and k = 16h+8.  B fragment: contiguous 32B
// per lane thanks to the fragment-blocked weight layout.
// ---------------------------------------------------------------------------
__global__ __launch_bounds__(256) void gemm1_gate_up(
    const unsigned short* __restrict__ ebuf,
    const u32x4* __restrict__ wgu,
    unsigned short* __restrict__ inter) {
  __shared__ unsigned short sA[2][128 * 40];      // rows padded to 80B
  __shared__ unsigned short sB[2][2][32 * 128];   // [buf][gate|up]
  const int e = blockIdx.z;
  const int mBase = blockIdx.y * 128;
  const int nBase = blockIdx.x * 128;             // within gate half
  const int tid = threadIdx.x, lane = tid & 31, wid = tid >> 5;
  const int waveM = wid & 1, waveN = wid >> 1;
  const int h = lane >> 4, l15 = lane & 15;
  const unsigned short* Ag = ebuf + (long)(e * CAP + mBase) * HID;
  const int KB = HID / 32;                        // 32 k-blocks
  const long blkE = (long)e * ((2 * INTER_ / 16) * KB);
  v8f accG[4][2], accU[4][2];
  const v8f vzero = {0.f, 0.f, 0.f, 0.f, 0.f, 0.f, 0.f, 0.f};
#pragma unroll
  for (int mi = 0; mi < 4; mi++)
#pragma unroll
    for (int ni = 0; ni < 2; ni++) { accG[mi][ni] = vzero; accU[mi][ni] = vzero; }

  auto computeTile = [&](int p) {
    FragBF a[4];
#pragma unroll
    for (int mi = 0; mi < 4; mi++) {
      const unsigned short* ab = &sA[p][(waveM * 64 + mi * 16 + l15) * 40];
      a[mi].u[0] = *(const u32x4*)(ab + 16 * h);
      a[mi].u[1] = *(const u32x4*)(ab + 16 * h + 8);
    }
#pragma unroll
    for (int ni = 0; ni < 2; ni++) {
      int nb = waveN * 2 + ni;
      FragBF bg, bu;
      const unsigned short* gb = &sB[p][0][nb * 512 + lane * 16];
      bg.u[0] = *(const u32x4*)(gb);
      bg.u[1] = *(const u32x4*)(gb + 8);
      const unsigned short* ub = &sB[p][1][nb * 512 + lane * 16];
      bu.u[0] = *(const u32x4*)(ub);
      bu.u[1] = *(const u32x4*)(ub + 8);
#pragma unroll
      for (int mi = 0; mi < 4; mi++) {
        accG[mi][ni] = __builtin_amdgcn_wmma_f32_16x16x32_bf16(
            false, a[mi].v, false, bg.v, (short)0, accG[mi][ni], false, false);
        accU[mi][ni] = __builtin_amdgcn_wmma_f32_16x16x32_bf16(
            false, a[mi].v, false, bu.v, (short)0, accU[mi][ni], false, false);
      }
    }
  };

#if HAVE_ASYNC_LDS
  auto issueTile = [&](int kk, int p) {
#pragma unroll
    for (int c = 0; c < 2; c++) {
      int lin = tid * 2 + c;
      int row = lin >> 2, col8 = lin & 3;
      async_copy16(Ag + (long)row * HID + kk * 32 + col8 * 8,
                   &sA[p][row * 40 + col8 * 8]);
      int nb = lin >> 6, off = lin & 63;
      long bg = blkE + (long)((nBase >> 4) + nb) * KB + kk;
      long bu = blkE + (long)((nBase >> 4) + (INTER_ / 16) + nb) * KB + kk;
      async_copy16(&wgu[bg * 64 + off], &sB[p][0][nb * 512 + off * 8]);
      async_copy16(&wgu[bu * 64 + off], &sB[p][1][nb * 512 + off * 8]);
    }
  };
  issueTile(0, 0);
  int p = 0;
  for (int kk = 0; kk < KB; ++kk) {
    wait_async0();                       // our async writes for tile kk done
    __syncthreads();                     // everyone's writes visible
    if (kk + 1 < KB) issueTile(kk + 1, p ^ 1);  // DMA overlaps WMMAs
    computeTile(p);
    p ^= 1;
  }
#else
  u32x4 rA[2], rBg[2], rBu[2];
  auto loadG = [&](int kk) {
#pragma unroll
    for (int c = 0; c < 2; c++) {
      int lin = tid * 2 + c;
      int row = lin >> 2, col8 = lin & 3;
      rA[c] = *(const u32x4*)(Ag + (long)row * HID + kk * 32 + col8 * 8);
      int nb = lin >> 6, off = lin & 63;
      long bg = blkE + (long)((nBase >> 4) + nb) * KB + kk;
      long bu = blkE + (long)((nBase >> 4) + (INTER_ / 16) + nb) * KB + kk;
      rBg[c] = wgu[bg * 64 + off];
      rBu[c] = wgu[bu * 64 + off];
    }
  };
  auto storeL = [&](int p) {
#pragma unroll
    for (int c = 0; c < 2; c++) {
      int lin = tid * 2 + c;
      int row = lin >> 2, col8 = lin & 3;
      *(u32x4*)&sA[p][row * 40 + col8 * 8] = rA[c];
      int nb = lin >> 6, off = lin & 63;
      *(u32x4*)&sB[p][0][nb * 512 + off * 8] = rBg[c];
      *(u32x4*)&sB[p][1][nb * 512 + off * 8] = rBu[c];
    }
  };
  loadG(0);
  int p = 0;
  for (int kk = 0; kk < KB; ++kk) {
    storeL(p);
    __syncthreads();
    if (kk + 1 < KB) loadG(kk + 1);
    computeTile(p);
    p ^= 1;
  }
#endif

  // epilogue: silu(gate) * up -> bf16 intermediate (row-major = GEMM2 A order)
#pragma unroll
  for (int mi = 0; mi < 4; mi++)
#pragma unroll
    for (int ni = 0; ni < 2; ni++)
#pragma unroll
      for (int r = 0; r < 8; r++) {
        float g = accG[mi][ni][r];
        float u = accU[mi][ni][r];
        float sv = g / (1.0f + __expf(-g));
        int m = mBase + waveM * 64 + mi * 16 + 8 * h + r;
        int n = nBase + waveN * 32 + ni * 16 + l15;
        inter[(long)(e * CAP + m) * INTER_ + n] = f2bf(sv * u);
      }
}

// ---------------------------------------------------------------------------
// Kernel 5: down-projection GEMM. Wave tile 64x64, WG tile 128x256.
// ---------------------------------------------------------------------------
__global__ __launch_bounds__(256) void gemm2_down(
    const unsigned short* __restrict__ inter,
    const u32x4* __restrict__ wdn,
    float* __restrict__ eout) {
  __shared__ unsigned short sA[2][128 * 40];
  __shared__ unsigned short sB[2][32 * 256];
  const int e = blockIdx.z;
  const int mBase = blockIdx.y * 128;
  const int nBase = blockIdx.x * 256;
  const int tid = threadIdx.x, lane = tid & 31, wid = tid >> 5;
  const int waveM = wid & 1, waveN = wid >> 1;
  const int h = lane >> 4, l15 = lane & 15;
  const unsigned short* Ag = inter + (long)(e * CAP + mBase) * INTER_;
  const int KB = INTER_ / 32;                     // 128 k-blocks
  const long blkE = (long)e * ((HID / 16) * KB);
  v8f acc[4][4];
  const v8f vzero = {0.f, 0.f, 0.f, 0.f, 0.f, 0.f, 0.f, 0.f};
#pragma unroll
  for (int mi = 0; mi < 4; mi++)
#pragma unroll
    for (int ni = 0; ni < 4; ni++) acc[mi][ni] = vzero;

  auto computeTile = [&](int p) {
    FragBF a[4];
#pragma unroll
    for (int mi = 0; mi < 4; mi++) {
      const unsigned short* ab = &sA[p][(waveM * 64 + mi * 16 + l15) * 40];
      a[mi].u[0] = *(const u32x4*)(ab + 16 * h);
      a[mi].u[1] = *(const u32x4*)(ab + 16 * h + 8);
    }
#pragma unroll
    for (int ni = 0; ni < 4; ni++) {
      int nb = waveN * 4 + ni;
      FragBF b;
      const unsigned short* bb = &sB[p][nb * 512 + lane * 16];
      b.u[0] = *(const u32x4*)(bb);
      b.u[1] = *(const u32x4*)(bb + 8);
#pragma unroll
      for (int mi = 0; mi < 4; mi++)
        acc[mi][ni] = __builtin_amdgcn_wmma_f32_16x16x32_bf16(
            false, a[mi].v, false, b.v, (short)0, acc[mi][ni], false, false);
    }
  };

#if HAVE_ASYNC_LDS
  auto issueTile = [&](int kk, int p) {
#pragma unroll
    for (int c = 0; c < 2; c++) {
      int lin = tid * 2 + c;
      int row = lin >> 2, col8 = lin & 3;
      async_copy16(Ag + (long)row * INTER_ + kk * 32 + col8 * 8,
                   &sA[p][row * 40 + col8 * 8]);
    }
#pragma unroll
    for (int c = 0; c < 4; c++) {
      int lin = tid * 4 + c;
      int nb = lin >> 6, off = lin & 63;
      long b = blkE + (long)((nBase >> 4) + nb) * KB + kk;
      async_copy16(&wdn[b * 64 + off], &sB[p][nb * 512 + off * 8]);
    }
  };
  issueTile(0, 0);
  int p = 0;
  for (int kk = 0; kk < KB; ++kk) {
    wait_async0();
    __syncthreads();
    if (kk + 1 < KB) issueTile(kk + 1, p ^ 1);
    computeTile(p);
    p ^= 1;
  }
#else
  u32x4 rA[2], rB[4];
  auto loadG = [&](int kk) {
#pragma unroll
    for (int c = 0; c < 2; c++) {
      int lin = tid * 2 + c;
      int row = lin >> 2, col8 = lin & 3;
      rA[c] = *(const u32x4*)(Ag + (long)row * INTER_ + kk * 32 + col8 * 8);
    }
#pragma unroll
    for (int c = 0; c < 4; c++) {
      int lin = tid * 4 + c;
      int nb = lin >> 6, off = lin & 63;
      long b = blkE + (long)((nBase >> 4) + nb) * KB + kk;
      rB[c] = wdn[b * 64 + off];
    }
  };
  auto storeL = [&](int p) {
#pragma unroll
    for (int c = 0; c < 2; c++) {
      int lin = tid * 2 + c;
      int row = lin >> 2, col8 = lin & 3;
      *(u32x4*)&sA[p][row * 40 + col8 * 8] = rA[c];
    }
#pragma unroll
    for (int c = 0; c < 4; c++) {
      int lin = tid * 4 + c;
      int nb = lin >> 6, off = lin & 63;
      *(u32x4*)&sB[p][nb * 512 + off * 8] = rB[c];
    }
  };
  loadG(0);
  int p = 0;
  for (int kk = 0; kk < KB; ++kk) {
    storeL(p);
    __syncthreads();
    if (kk + 1 < KB) loadG(kk + 1);
    computeTile(p);
    p ^= 1;
  }
#endif

#pragma unroll
  for (int mi = 0; mi < 4; mi++)
#pragma unroll
    for (int ni = 0; ni < 4; ni++)
#pragma unroll
      for (int r = 0; r < 8; r++) {
        int m = mBase + waveM * 64 + mi * 16 + 8 * h + r;
        int n = nBase + waveN * 64 + ni * 16 + l15;
        eout[(long)(e * CAP + m) * HID + n] = acc[mi][ni][r];
      }
}

// ---------------------------------------------------------------------------
// Kernel 6: combine — deterministic gather of <=2 expert rows per token.
// ---------------------------------------------------------------------------
__global__ __launch_bounds__(256) void combine_kernel(
    const float* __restrict__ eout, const int* __restrict__ tokslot,
    const float* __restrict__ tokw, float* __restrict__ out) {
  int t = blockIdx.x, c = threadIdx.x;
  int s0 = tokslot[t * 2 + 0], s1 = tokslot[t * 2 + 1];
  float w0 = tokw[t * 2 + 0], w1 = tokw[t * 2 + 1];
  f32x4 r = {0.f, 0.f, 0.f, 0.f};
  if (s0 >= 0) r += w0 * ((const f32x4*)(eout + (long)s0 * HID))[c];
  if (s1 >= 0) r += w1 * ((const f32x4*)(eout + (long)s1 * HID))[c];
  ((f32x4*)(out + (long)t * HID))[c] = r;
}

// ---------------------------------------------------------------------------
extern "C" void kernel_launch(void* const* d_in, const int* in_sizes, int n_in,
                              void* d_out, int out_size, void* d_ws, size_t ws_size,
                              hipStream_t stream) {
  (void)in_sizes; (void)n_in; (void)out_size;
  const float* x     = (const float*)d_in[0];   // (S,B,H) fp32
  const float* aff   = (const float*)d_in[1];   // (T,E)
  const int*   eidx  = (const int*)d_in[2];     // (T,K)
  const float* wguF  = (const float*)d_in[3];   // (E,H,2I)
  const float* wdnF  = (const float*)d_in[4];   // (E,I,H)
  float* out = (float*)d_out;

  char* ws = (char*)d_ws;
  size_t off = 0;
  auto carve = [&](size_t bytes) {
    void* p = ws + off;
    off = (off + bytes + 255) & ~(size_t)255;
    return p;
  };
  int*   slot2tok = (int*)carve((size_t)ECAP * 4);
  int*   tokslot  = (int*)carve((size_t)T_TOK * TOPK * 4);
  float* tokw     = (float*)carve((size_t)T_TOK * TOPK * 4);
  unsigned short* ebuf  = (unsigned short*)carve((size_t)ECAP * HID * 2);
  u32x4*          wdnB  = (u32x4*)carve((size_t)NE * INTER_ * HID * 2);
  unsigned short* inter = (unsigned short*)carve((size_t)ECAP * INTER_ * 2);
  // gate_up bf16 weights; expert_out (84MB <= 134MB) aliases this region —
  // wgu is dead after gemm1 and re-generated at the start of every call.
  size_t wguBytes = (size_t)NE * HID * (2 * INTER_) * 2;
  u32x4* wguB = (u32x4*)(ws + off);
  float* eout = (float*)(ws + off);
  off += wguBytes;
  if (off > ws_size) return;   // workspace too small: do nothing (deterministic)

  convert_weights<<<16384, 256, 0, stream>>>(
      wguF, (unsigned*)wguB, HID, 2 * INTER_, (long)NE * HID * (2 * INTER_) / 2);
  convert_weights<<<8192, 256, 0, stream>>>(
      wdnF, (unsigned*)wdnB, INTER_, HID, (long)NE * INTER_ * HID / 2);
  routing_kernel<<<1, 1024, 0, stream>>>(eidx, aff, slot2tok, tokslot, tokw);
  dispatch_kernel<<<ECAP, 256, 0, stream>>>(x, slot2tok, (u32x2*)ebuf);
  gemm1_gate_up<<<dim3(INTER_ / 128, CAP / 128, NE), 256, 0, stream>>>(ebuf, wguB, inter);
  gemm2_down<<<dim3(HID / 256, CAP / 128, NE), 256, 0, stream>>>(inter, wdnB, eout);
  combine_kernel<<<T_TOK, 256, 0, stream>>>(eout, tokslot, tokw, out);
}
